// Mamba2_46059229283141
// MI455X (gfx1250) — compile-verified, matrix operand order checked by reading, not run on hardware
//
#include <hip/hip_runtime.h>

// ---------------- problem constants (match reference) ----------------
#define D_MODEL   1024
#define D_INNER   2048
#define HEADDIM   64
#define NHEADS    32
#define D_STATE   128
#define D_CONV    4
#define CONV_DIM  (D_INNER + 2 * D_STATE)                 // 2304
#define D_IN_PROJ (2 * D_INNER + 2 * D_STATE + NHEADS)    // 4384
#define N1P       4480                                    // D_IN_PROJ padded to /128
#define BATCH     2
#define SEQLEN    2048
#define BL        (BATCH * SEQLEN)                        // 4096
#define EPS_RMS   1e-5f
#define Q         64                                      // SSD chunk length
#define NCHUNK    (SEQLEN / Q)                            // 32

typedef __attribute__((ext_vector_type(16))) __bf16 v16bf;
typedef __attribute__((ext_vector_type(8)))  __bf16 v8bf;
typedef __attribute__((ext_vector_type(8)))  float  v8f;

// ---------------- helpers ----------------
__device__ __forceinline__ unsigned short f2bf_rne(float f) {
  unsigned int x = __float_as_uint(f);
  x += 0x7FFFu + ((x >> 16) & 1u);        // round-to-nearest-even
  return (unsigned short)(x >> 16);
}

__device__ __forceinline__ float silu_f(float v) {
  return v / (1.0f + __expf(-v));
}

// A/B fragment gather per CDNA5 16-bit 16x32 operand layout:
//   lanes 0-15  hold K = {k0..k0+7, k0+16..k0+23}
//   lanes 16-31 hold K = {k0+8..k0+15, k0+24..k0+31}   (kph = 8)
// Works for global or LDS (generic pointer); two 16B loads.
__device__ __forceinline__ v16bf load_frag(const unsigned short* base, int row,
                                           int stride, int k0, int kph) {
  const v8bf* p = (const v8bf*)(base + (size_t)row * stride + k0 + kph);
  return __builtin_shufflevector(p[0], p[2],
      0,1,2,3,4,5,6,7,8,9,10,11,12,13,14,15);
}

__device__ __forceinline__ v8f bwmma(v16bf a, v16bf b, v8f c) {
  return __builtin_amdgcn_wmma_f32_16x16x32_bf16(
      false, a, false, b, (short)0, c, false, false);
}

// scale every element of a bf16 fragment by a per-row scalar (A rows are
// per-lane uniform, so one scalar per lane covers the whole fragment)
__device__ __forceinline__ v16bf scale_frag(v16bf a, float s) {
  v16bf r;
#pragma unroll
  for (int i = 0; i < 16; ++i) r[i] = (__bf16)((float)a[i] * s);
  return r;
}

// ---------------- cast kernels (f32 -> bf16 operands for WMMA) ----------------
__global__ void cast_bf16_kernel(const float* __restrict__ src,
                                 unsigned short* __restrict__ dst, long n) {
  long i = (long)blockIdx.x * blockDim.x + threadIdx.x;
  long stride = (long)gridDim.x * blockDim.x;
  for (; i < n; i += stride) dst[i] = f2bf_rne(src[i]);
}

__global__ void cast_pad_bf16_kernel(const float* __restrict__ src,
                                     unsigned short* __restrict__ dst,
                                     int rows_valid, long n_total, int K) {
  long i = (long)blockIdx.x * blockDim.x + threadIdx.x;
  long stride = (long)gridDim.x * blockDim.x;
  for (; i < n_total; i += stride) {
    long row = i / K;
    dst[i] = (row < rows_valid) ? f2bf_rne(src[i]) : (unsigned short)0;
  }
}

// ---------------- WMMA bf16 GEMM:  C(MxN,f32) = A(MxK,bf16) * B(NxK,bf16)^T ----
// Block: 256 threads = 8 waves as 4(M) x 2(N).  Block tile: 64(M) x 128(N).
__global__ __launch_bounds__(256) void wmma_gemm_bf16_nt(
    const unsigned short* __restrict__ A,   // M x K, row-major bf16
    const unsigned short* __restrict__ B,   // Npad x K, row-major bf16
    float* __restrict__ C,                  // M x ldc, f32
    int K, int ldc) {
  const int lane = threadIdx.x & 31;
  const int wave = threadIdx.x >> 5;
  const int wm   = wave & 3;
  const int wn   = wave >> 2;
  const int m0   = blockIdx.y * 64 + wm * 16;
  const int n0   = blockIdx.x * 128 + wn * 64;
  const int lr   = lane & 15;
  const int kph  = (lane >> 4) << 3;

  const unsigned short* arow = A + (size_t)(m0 + lr) * K;
  v8f acc[4] = {};

  for (int k0 = 0; k0 < K; k0 += 32) {
    v16bf af = load_frag(arow, 0, 0, k0, kph);
#pragma unroll
    for (int t = 0; t < 4; ++t) {
      v16bf bf = load_frag(B, n0 + t * 16 + lr, K, k0, kph);
      acc[t] = bwmma(af, bf, acc[t]);
    }
  }

  const int mrow = m0 + ((lane >> 4) << 3);
#pragma unroll
  for (int t = 0; t < 4; ++t) {
    float* cp = C + (size_t)mrow * ldc + (n0 + t * 16 + lr);
#pragma unroll
    for (int r = 0; r < 8; ++r) cp[(size_t)r * ldc] = acc[t][r];
  }
}

// ---------------- causal depthwise conv (K=4) + SiLU over xBC slice ----------
__global__ void conv_silu_kernel(const float* __restrict__ zx,   // BL x N1P
                                 const float* __restrict__ cw,   // CONV_DIM x 4
                                 const float* __restrict__ cb,   // CONV_DIM
                                 float* __restrict__ xbc) {      // BL x CONV_DIM
  long i = (long)blockIdx.x * blockDim.x + threadIdx.x;
  if (i >= (long)BL * CONV_DIM) return;
  int c   = (int)(i % CONV_DIM);
  int row = (int)(i / CONV_DIM);
  int l   = row & (SEQLEN - 1);
  float acc = cb[c];
#pragma unroll
  for (int k = 0; k < D_CONV; ++k) {
    int dl = l - (D_CONV - 1) + k;
    if (dl >= 0)
      acc += zx[(size_t)(row - (D_CONV - 1) + k) * N1P + D_INNER + c] * cw[c * D_CONV + k];
  }
  xbc[(size_t)row * CONV_DIM + c] = silu_f(acc);
}

// -------- dt = softplus(dt_raw + bias); ldA = dt * (-exp(A_log)) -------------
__global__ void dt_kernel(const float* __restrict__ zx,
                          const float* __restrict__ dt_bias,
                          const float* __restrict__ A_log,
                          float* __restrict__ dt_out,
                          float* __restrict__ lda_out) {
  int i = blockIdx.x * blockDim.x + threadIdx.x;
  if (i >= BL * NHEADS) return;
  int h   = i & (NHEADS - 1);
  int row = i >> 5;
  float v  = zx[(size_t)row * N1P + D_INNER + CONV_DIM + h] + dt_bias[h];
  float dt = (v > 20.0f) ? v : log1pf(__expf(v));
  float A  = -__expf(A_log[h]);
  dt_out[i]  = dt;
  lda_out[i] = dt * A;
}

// ===================== chunked SSD =====================
// Kernel A: per (chunk, b*h) — cumsum of log-decay; chunk state
//   S[p,n] = sum_t (dt_t * exp(cum[Q-1]-cum[t]) * x[t,p]) * B[t,n]
// via WMMA GEMM  (M=64 p, N=128 n, K=64 t).
__global__ __launch_bounds__(256) void chunk_state_kernel(
    const float* __restrict__ xbc, const float* __restrict__ dtb,
    const float* __restrict__ lda, float* __restrict__ Sbuf,
    float* __restrict__ gcum) {
  const int c  = blockIdx.x;
  const int bh = blockIdx.y;
  const int b  = bh >> 5;
  const int h  = bh & (NHEADS - 1);
  const int row0 = b * SEQLEN + c * Q;
  const size_t bhc = (size_t)bh * NCHUNK + c;

  const int tid  = threadIdx.x;
  const int lane = tid & 31;
  const int wave = tid >> 5;
  const int lr   = lane & 15;
  const int kph  = (lane >> 4) << 3;
  const int off8 = (lane >> 4) << 3;

  __shared__ __align__(16) unsigned short sXwT[Q * Q];       // [p][t], weighted x
  __shared__ __align__(16) unsigned short sBT[D_STATE * Q];  // [n][t]
  __shared__ float sCumF[Q];
  __shared__ float sDtS[Q];
  __shared__ float sW[Q];

  if (tid < Q) {
    sDtS[tid]  = dtb[(size_t)(row0 + tid) * NHEADS + h];
    sCumF[tid] = lda[(size_t)(row0 + tid) * NHEADS + h];
  }
  __syncthreads();
  // inclusive Hillis-Steele scan over Q=64 log-decays
  for (int s = 1; s < Q; s <<= 1) {
    float v = 0.0f;
    if (tid < Q && tid >= s) v = sCumF[tid - s];
    __syncthreads();
    if (tid < Q) sCumF[tid] += v;
    __syncthreads();
  }
  if (tid < Q) {
    sW[tid] = sDtS[tid] * __expf(sCumF[Q - 1] - sCumF[tid]);  // <= dt (safe)
    gcum[bhc * Q + tid] = sCumF[tid];
  }
  __syncthreads();

  for (int i = tid; i < Q * Q; i += 256) {
    int t = i >> 6, p = i & (Q - 1);
    float xv = xbc[(size_t)(row0 + t) * CONV_DIM + h * HEADDIM + p];
    sXwT[p * Q + t] = f2bf_rne(xv * sW[t]);
  }
  for (int i = tid; i < Q * D_STATE; i += 256) {
    int t = i >> 7, n = i & (D_STATE - 1);
    sBT[n * Q + t] = f2bf_rne(xbc[(size_t)(row0 + t) * CONV_DIM + D_INNER + n]);
  }
  __syncthreads();

  // 32 tiles (4 p-tiles x 8 n-tiles), 4 per wave; K = 64 (two wmma steps)
  v8f acc[4] = {};
#pragma unroll
  for (int k0 = 0; k0 < Q; k0 += 32) {
#pragma unroll
    for (int q2 = 0; q2 < 4; ++q2) {
      int tile = wave * 4 + q2;
      int pm = (tile >> 3) << 4, nn = (tile & 7) << 4;
      v16bf af = load_frag(sXwT, pm + lr, Q, k0, kph);
      v16bf bf = load_frag(sBT,  nn + lr, Q, k0, kph);
      acc[q2] = bwmma(af, bf, acc[q2]);
    }
  }
  float* Sp = Sbuf + bhc * (size_t)(HEADDIM * D_STATE);
#pragma unroll
  for (int q2 = 0; q2 < 4; ++q2) {
    int tile = wave * 4 + q2;
    int pm = (tile >> 3) << 4, nn = (tile & 7) << 4;
#pragma unroll
    for (int r = 0; r < 8; ++r)
      Sp[(size_t)(pm + off8 + r) * D_STATE + (nn + lr)] = acc[q2][r];
  }
}

// Kernel B: inter-chunk recurrence  H <- H * exp(cum[Q-1]) + S_c, per (b*h).
// Elementwise in (p,n): 32 sequential FMAs per thread, no barriers.
__global__ __launch_bounds__(256) void chunk_recur_kernel(
    const float* __restrict__ Sbuf, const float* __restrict__ gcum,
    unsigned short* __restrict__ Hin) {
  const int bh  = blockIdx.x;
  const int tid = threadIdx.x;
  float hst[32];
#pragma unroll
  for (int j = 0; j < 32; ++j) hst[j] = 0.0f;

  for (int c = 0; c < NCHUNK; ++c) {
    const size_t bhc  = (size_t)bh * NCHUNK + c;
    const size_t base = bhc * (size_t)(HEADDIM * D_STATE);
    const float  lam  = __expf(gcum[bhc * Q + (Q - 1)]);
#pragma unroll
    for (int j = 0; j < 32; ++j) {
      size_t idx = base + (size_t)j * 256 + tid;
      Hin[idx] = f2bf_rne(hst[j]);                 // state entering chunk c
      hst[j]   = hst[j] * lam + Sbuf[idx];
    }
  }
}

// Kernel C: per (chunk, b*h) — chunk outputs via WMMA:
//   G0 = C @ B^T (QxQ over D_STATE); G = mask(G0)*exp(cum[t]-cum[s])*dt[s]
//   Y  = G @ X^T  +  (C * exp(cum[t])) @ Hin^T  + D*x
__global__ __launch_bounds__(256) void chunk_out_kernel(
    const float* __restrict__ xbc, const float* __restrict__ dtb,
    const float* __restrict__ gcum, const unsigned short* __restrict__ Hin,
    const float* __restrict__ Dp, float* __restrict__ y) {
  const int c  = blockIdx.x;
  const int bh = blockIdx.y;
  const int b  = bh >> 5;
  const int h  = bh & (NHEADS - 1);
  const int row0 = b * SEQLEN + c * Q;
  const size_t bhc = (size_t)bh * NCHUNK + c;

  const int tid  = threadIdx.x;
  const int lane = tid & 31;
  const int wave = tid >> 5;
  const int lr   = lane & 15;
  const int kph  = (lane >> 4) << 3;
  const int off8 = (lane >> 4) << 3;

  __shared__ __align__(16) unsigned short sC[Q * D_STATE];   // [t][n]
  __shared__ __align__(16) unsigned short sBH[Q * D_STATE];  // B [s][n], then Hin [p][n]
  __shared__ __align__(16) unsigned short sXT[Q * Q];        // [p][t]
  __shared__ __align__(16) unsigned short sG[Q * Q];         // [t][s]
  __shared__ float sCumF[Q];
  __shared__ float sDtS[Q];

  if (tid < Q) {
    sCumF[tid] = gcum[bhc * Q + tid];
    sDtS[tid]  = dtb[(size_t)(row0 + tid) * NHEADS + h];
  }
  __syncthreads();

  for (int i = tid; i < Q * D_STATE; i += 256) {
    int t = i >> 7, n = i & (D_STATE - 1);
    const float* rp = xbc + (size_t)(row0 + t) * CONV_DIM + D_INNER;
    sC[i]  = f2bf_rne(rp[D_STATE + n]);   // C slice
    sBH[i] = f2bf_rne(rp[n]);             // B slice
  }
  for (int i = tid; i < Q * Q; i += 256) {
    int t = i >> 6, p = i & (Q - 1);
    sXT[p * Q + t] = f2bf_rne(xbc[(size_t)(row0 + t) * CONV_DIM + h * HEADDIM + p]);
  }
  __syncthreads();

  // ---- G0 = C @ B^T, masked causal decay -> sG (bf16) ----
  {
    v8f acc[2] = {};
#pragma unroll
    for (int k0 = 0; k0 < D_STATE; k0 += 32) {
#pragma unroll
      for (int q2 = 0; q2 < 2; ++q2) {
        int tile = wave * 2 + q2;
        int tm = (tile >> 2) << 4, sn = (tile & 3) << 4;
        v16bf af = load_frag(sC,  tm + lr, D_STATE, k0, kph);
        v16bf bf = load_frag(sBH, sn + lr, D_STATE, k0, kph);
        acc[q2] = bwmma(af, bf, acc[q2]);
      }
    }
#pragma unroll
    for (int q2 = 0; q2 < 2; ++q2) {
      int tile = wave * 2 + q2;
      int tm = (tile >> 2) << 4, sn = (tile & 3) << 4;
      int s_ = sn + lr;
#pragma unroll
      for (int r = 0; r < 8; ++r) {
        int t_ = tm + off8 + r;
        float g = (s_ <= t_)
            ? acc[q2][r] * __expf(sCumF[t_] - sCumF[s_]) * sDtS[s_] : 0.0f;
        sG[t_ * Q + s_] = f2bf_rne(g);
      }
    }
  }
  __syncthreads();

  // restage sBH <- Hin (bf16 passthrough)
  {
    const unsigned short* hp = Hin + bhc * (size_t)(HEADDIM * D_STATE);
    for (int i = tid; i < Q * D_STATE; i += 256) sBH[i] = hp[i];
  }
  __syncthreads();

  // ---- Y = G @ X^T + (C * exp(cum)) @ Hin^T ----
  v8f yacc[2] = {};
#pragma unroll
  for (int k0 = 0; k0 < Q; k0 += 32) {       // intra-chunk
#pragma unroll
    for (int q2 = 0; q2 < 2; ++q2) {
      int tile = wave * 2 + q2;
      int tm = (tile >> 2) << 4, pn = (tile & 3) << 4;
      v16bf af = load_frag(sG,  tm + lr, Q, k0, kph);
      v16bf bf = load_frag(sXT, pn + lr, Q, k0, kph);
      yacc[q2] = bwmma(af, bf, yacc[q2]);
    }
  }
#pragma unroll
  for (int k0 = 0; k0 < D_STATE; k0 += 32) { // inter-chunk
#pragma unroll
    for (int q2 = 0; q2 < 2; ++q2) {
      int tile = wave * 2 + q2;
      int tm = (tile >> 2) << 4, pn = (tile & 3) << 4;
      float cs = __expf(sCumF[tm + lr]);     // per-row scale (<= 1)
      v16bf af = scale_frag(load_frag(sC, tm + lr, D_STATE, k0, kph), cs);
      v16bf bf = load_frag(sBH, pn + lr, D_STATE, k0, kph);
      yacc[q2] = bwmma(af, bf, yacc[q2]);
    }
  }

  const float Dh = Dp[h];
#pragma unroll
  for (int q2 = 0; q2 < 2; ++q2) {
    int tile = wave * 2 + q2;
    int tm = (tile >> 2) << 4, pn = (tile & 3) << 4;
    int p_ = pn + lr;
#pragma unroll
    for (int r = 0; r < 8; ++r) {
      int t_ = tm + off8 + r;
      int row = row0 + t_;
      float xv = xbc[(size_t)row * CONV_DIM + h * HEADDIM + p_];
      y[(size_t)row * D_INNER + h * HEADDIM + p_] = yacc[q2][r] + Dh * xv;
    }
  }
}

// ---------------- y *= silu(z); RMS-norm * norm_w; cast to bf16 --------------
__global__ __launch_bounds__(256) void gate_norm_kernel(
    const float* __restrict__ y,       // BL x D_INNER
    const float* __restrict__ zx,      // BL x N1P (z = cols [0, D_INNER))
    const float* __restrict__ norm_w,  // D_INNER
    unsigned short* __restrict__ y_bf) {
  const int row = blockIdx.x;
  const int t   = threadIdx.x;
  float local[8];
  float ss = 0.0f;
#pragma unroll
  for (int i = 0; i < 8; ++i) {
    int c = t + i * 256;
    float z = zx[(size_t)row * N1P + c];
    float v = y[(size_t)row * D_INNER + c] * silu_f(z);
    local[i] = v;
    ss += v * v;
  }
  __shared__ float red[256];
  red[t] = ss;
  __syncthreads();
  for (int s = 128; s > 0; s >>= 1) {
    if (t < s) red[t] += red[t + s];
    __syncthreads();
  }
  const float r = rsqrtf(red[0] * (1.0f / (float)D_INNER) + EPS_RMS);
#pragma unroll
  for (int i = 0; i < 8; ++i) {
    int c = t + i * 256;
    y_bf[(size_t)row * D_INNER + c] = f2bf_rne(local[i] * r * norm_w[c]);
  }
}

// ---------------- host launcher ----------------
extern "C" void kernel_launch(void* const* d_in, const int* in_sizes, int n_in,
                              void* d_out, int out_size, void* d_ws, size_t ws_size,
                              hipStream_t stream) {
  (void)in_sizes; (void)n_in; (void)out_size; (void)ws_size;
  const float* u       = (const float*)d_in[0];
  const float* W_in    = (const float*)d_in[1];
  const float* conv_w  = (const float*)d_in[2];
  const float* conv_b  = (const float*)d_in[3];
  const float* dt_bias = (const float*)d_in[4];
  const float* A_log   = (const float*)d_in[5];
  const float* Dp      = (const float*)d_in[6];
  const float* norm_w  = (const float*)d_in[7];
  const float* W_out   = (const float*)d_in[8];
  float* out = (float*)d_out;

  char* ws = (char*)d_ws;
  size_t off = 0;
  auto take = [&](size_t bytes) -> char* {
    char* p = ws + off;
    off = (off + bytes + 255) & ~(size_t)255;
    return p;
  };
  unsigned short* u_bf    = (unsigned short*)take((size_t)BL * D_MODEL * 2);
  unsigned short* Win_bf  = (unsigned short*)take((size_t)N1P * D_MODEL * 2);
  unsigned short* Wout_bf = (unsigned short*)take((size_t)D_MODEL * D_INNER * 2);
  float* zx    = (float*)take((size_t)BL * N1P * 4);
  float* xbc   = (float*)take((size_t)BL * CONV_DIM * 4);
  float* dtb   = (float*)take((size_t)BL * NHEADS * 4);
  float* ldab  = (float*)take((size_t)BL * NHEADS * 4);
  float* gcum  = (float*)take((size_t)BATCH * NHEADS * NCHUNK * Q * 4);
  float* Sbuf  = (float*)take((size_t)BATCH * NHEADS * NCHUNK * HEADDIM * D_STATE * 4);
  unsigned short* Hin = (unsigned short*)take((size_t)BATCH * NHEADS * NCHUNK * HEADDIM * D_STATE * 2);
  float* y     = (float*)take((size_t)BL * D_INNER * 4);
  unsigned short* y_bf = (unsigned short*)take((size_t)BL * D_INNER * 2);

  // operand casts
  cast_bf16_kernel<<<2048, 256, 0, stream>>>(u, u_bf, (long)BL * D_MODEL);
  cast_pad_bf16_kernel<<<2048, 256, 0, stream>>>(W_in, Win_bf, D_IN_PROJ,
                                                 (long)N1P * D_MODEL, D_MODEL);
  cast_bf16_kernel<<<2048, 256, 0, stream>>>(W_out, Wout_bf, (long)D_MODEL * D_INNER);

  // input projection: zx = u @ W_in^T   (M=4096, N=4480(pad), K=1024)
  {
    dim3 grid(N1P / 128, BL / 64);
    wmma_gemm_bf16_nt<<<grid, 256, 0, stream>>>(u_bf, Win_bf, zx, D_MODEL, N1P);
  }

  // conv + silu, dt/ldA
  conv_silu_kernel<<<(BL * CONV_DIM) / 256, 256, 0, stream>>>(zx, conv_w, conv_b, xbc);
  dt_kernel<<<(BL * NHEADS) / 256, 256, 0, stream>>>(zx, dt_bias, A_log, dtb, ldab);

  // chunked SSD
  {
    dim3 gc(NCHUNK, BATCH * NHEADS);
    chunk_state_kernel<<<gc, 256, 0, stream>>>(xbc, dtb, ldab, Sbuf, gcum);
    chunk_recur_kernel<<<BATCH * NHEADS, 256, 0, stream>>>(Sbuf, gcum, Hin);
    chunk_out_kernel<<<gc, 256, 0, stream>>>(xbc, dtb, gcum, Hin, Dp, y);
  }

  // gate + RMS norm -> bf16
  gate_norm_kernel<<<BL, 256, 0, stream>>>(y, zx, norm_w, y_bf);

  // output projection: out = y_n @ W_out^T  (M=4096, N=1024, K=2048)
  {
    dim3 grid(D_MODEL / 128, BL / 64);
    wmma_gemm_bf16_nt<<<grid, 256, 0, stream>>>(y_bf, Wout_bf, out, D_INNER, D_MODEL);
  }
}
